// PathFusionEmbedding_43353399885892
// MI455X (gfx1250) — matrix operand order repaired and checked by reading, not compiled, not run on hardware
//
#include <hip/hip_runtime.h>
#include <hip/hip_bf16.h>

// ---------------- problem constants ----------------
#define NTREES   10
#define NLEAVES  16
#define PLEN     5
#define NNODES   31
#define DD       256
#define BB       4096
#define NROWS    (BB * NTREES)      // 40960 independent LSTM rows
#define GDIM     (4 * DD)           // 1024 gate columns
#define NNROWS   (NTREES * NNODES)  // 310 distinct node-gate rows

// ---------------- LSTM kernel tiling ----------------
#define MROWS 64                     // rows per workgroup
#define HST   264                    // h LDS row stride (bf16 halves); 264 = 8*33, 16B-aligned rows

// workspace layout (d_ws)
#define PACKW_BYTES  (512 * 512 * 2)             // 512 fragments * 512 halves
#define NGATES_OFF   PACKW_BYTES                 // 524288
#define NGATES_BYTES (NNROWS * GDIM * 4)         // 1269760
#define NIDS_OFF     (NGATES_OFF + NGATES_BYTES) // 1794048

typedef __attribute__((ext_vector_type(16))) __bf16 v16bf;
typedef __attribute__((ext_vector_type(8)))  __bf16 v8bf;
typedef __attribute__((ext_vector_type(8)))  float  v8f;

__device__ __forceinline__ unsigned short f2bf(float f) {
  __bf16 b = (__bf16)f;
  return __builtin_bit_cast(unsigned short, b);
}
__device__ __forceinline__ float fast_rcp(float x) { return __builtin_amdgcn_rcpf(x); }
__device__ __forceinline__ float sigm(float x)  { return fast_rcp(1.f + __expf(-x)); }
__device__ __forceinline__ float tanh_f(float x){ float e = __expf(2.f * x); return (e - 1.f) * fast_rcp(e + 1.f); }

// -----------------------------------------------------------------------------
// Kernel 1: pre-swizzle W_hh (1024x256 f32, row-major) into bf16 WMMA B-fragments.
// We compute h @ W_hh^T, i.e. C=A*B with B[k][n] = W_hh[n][k].
// Fragment f = nt*8 + kt covers cols [nt*16,nt*16+16), k [kt*32,kt*32+32).
// 16-bit B layout used here (wave32, contiguous k, not ISA table order): lane l -> n = l&15; halves j=0..15 -> k = (l>>4)*16 + j.
// Stored so each lane's 16 halves (32B) are contiguous: packW[f*512 + l*16 + j].
// -----------------------------------------------------------------------------
__global__ void pack_whh_kernel(const float* __restrict__ Whh, unsigned short* __restrict__ packW) {
  int f  = blockIdx.x;                  // 0..511
  int nt = f >> 3, kt = f & 7;
  int l  = threadIdx.x;                 // 0..31 (one wave)
  int n  = nt * 16 + (l & 15);
  int k0 = kt * 32 + (l >> 4) * 16;
  const float* src = Whh + (size_t)n * DD + k0;
  unsigned short* dst = packW + (size_t)f * 512 + l * 16;
#pragma unroll
  for (int j = 0; j < 16; ++j) dst[j] = f2bf(src[j]);
}

// -----------------------------------------------------------------------------
// Kernel 2: leaf argmax -> 5 global node ids per row (row n = b*NTREES + t)
// -----------------------------------------------------------------------------
__global__ void leaf_paths_kernel(const float* __restrict__ cross, int* __restrict__ nodeids) {
  int idx = blockIdx.x * blockDim.x + threadIdx.x;
  if (idx >= NROWS) return;
  int b = idx / NTREES, t = idx - b * NTREES;
  const float* cf = cross + (size_t)b * (NTREES * NLEAVES) + t * NLEAVES;
  int leaf = 0; float best = cf[0];
#pragma unroll
  for (int l = 1; l < NLEAVES; ++l) { float v = cf[l]; if (v > best) { best = v; leaf = l; } }
  int base = t * NNODES;
  nodeids[idx * PLEN + 0] = base;
  nodeids[idx * PLEN + 1] = base + 1  + (leaf >> 3);
  nodeids[idx * PLEN + 2] = base + 3  + (leaf >> 2);
  nodeids[idx * PLEN + 3] = base + 7  + (leaf >> 1);
  nodeids[idx * PLEN + 4] = base + 15 + leaf;
}

// -----------------------------------------------------------------------------
// Kernel 3: node_gates[310][1024] = emb @ W_ih^T + (b_ih + b_hh)   (f32, tiny)
// grid = (8 col-slices of 128, 39 node groups of 8), block = 128 threads
// -----------------------------------------------------------------------------
__global__ void node_gates_kernel(const float* __restrict__ emb, const float* __restrict__ Wih,
                                  const float* __restrict__ bih, const float* __restrict__ bhh,
                                  float* __restrict__ ngates) {
  __shared__ float se[8][DD];
  int g  = blockIdx.x * 128 + threadIdx.x;   // gate column
  int n0 = blockIdx.y * 8;
  for (int i = threadIdx.x; i < 8 * DD; i += 128) {
    int nn = n0 + i / DD;
    se[i / DD][i % DD] = (nn < NNROWS) ? emb[(size_t)nn * DD + (i % DD)] : 0.f;
  }
  __syncthreads();
  float acc[8] = {0.f, 0.f, 0.f, 0.f, 0.f, 0.f, 0.f, 0.f};
  const float* w = Wih + (size_t)g * DD;
  for (int d = 0; d < DD; ++d) {
    float wv = w[d];
#pragma unroll
    for (int i = 0; i < 8; ++i) acc[i] += wv * se[i][d];
  }
  float bias = bih[g] + bhh[g];
#pragma unroll
  for (int i = 0; i < 8; ++i) {
    int nn = n0 + i;
    if (nn < NNROWS) ngates[(size_t)nn * GDIM + g] = acc[i] + bias;
  }
}

// -----------------------------------------------------------------------------
// Kernel 4: fused 5-step LSTM. One WG = 64 rows, 256 threads = 8 waves.
// Gate-interleaved column assignment: wave w owns, for EACH gate q in {i,f,g,o},
// the 32 columns [q*256 + w*32, q*256 + w*32 + 32). Accumulator tile n = q*2+j
// covers cols q*256 + w*32 + j*16. Hence every lane holds i/f/g/o for its own
// (row, d) elements in registers and the LSTM cell update runs directly on the
// WMMA accumulators -- no gate LDS round-trip. Only h (bf16, 34KB) lives in LDS.
// B fragments are software-pipelined 4 deep; A fragments double-buffered per kt.
// -----------------------------------------------------------------------------
__global__ void __launch_bounds__(256, 1)
lstm_fused_kernel(const unsigned short* __restrict__ packW,
                  const float* __restrict__ ngates,
                  const int*   __restrict__ nodeids,
                  float* __restrict__ out) {
  __shared__ unsigned short hls[MROWS * HST];   // h, bf16, row-major (padded stride)

  const int tid  = threadIdx.x;
  const int lane = tid & 31;
  const int wave = tid >> 5;
  const int row0 = blockIdx.x * MROWS;
  const int rhi  = (lane < 16) ? 0 : 8;         // C/D layout row offset for this lane
  const int dbase = wave * 32 + (lane & 15);    // within-gate column of this lane (j=0)

  // c[(m*2+j)*8 + r] = cell state for row m*16+rhi+r, col dbase + j*16
  float c[MROWS];
#pragma unroll
  for (int r = 0; r < MROWS; ++r) c[r] = 0.f;

  const v8f vzero = {0.f, 0.f, 0.f, 0.f, 0.f, 0.f, 0.f, 0.f};
  const unsigned short* pB0 = packW + lane * 16;
  const unsigned short* pA  = hls + (size_t)(lane & 15) * HST + (lane >> 4) * 16;

  // A fragment loader: 16-bit A 16x32, contiguous k (not ISA table order). lane l: M=(l&15);
  // halves j=0..15 -> k = kt*32 + (l>>4)*16 + j (matches packed B).
  auto loadA = [&](v16bf* dst, int kt) {
#pragma unroll
    for (int m = 0; m < 4; ++m) {
      const unsigned short* pa = pA + (size_t)(m * 16) * HST + kt * 32;
      v8bf lo = *(const v8bf*)pa;
      v8bf hi = *(const v8bf*)(pa + 8);
      dst[m] = __builtin_shufflevector(lo, hi, 0,1,2,3,4,5,6,7,8,9,10,11,12,13,14,15);
    }
  };
  // B fragment for accumulator tile n (= q*2+j) at K-chunk kt:
  // global 16-col tile index = q*16 + wave*2 + j
  auto loadB = [&](int kt, int n) -> v16bf {
    int ntg = (n >> 1) * 16 + wave * 2 + (n & 1);
    return *(const v16bf*)(pB0 + (size_t)(ntg * 8 + kt) * 512);
  };

#pragma unroll 1
  for (int t = 0; t < PLEN; ++t) {
    v8f acc[4][8];
#pragma unroll
    for (int m = 0; m < 4; ++m)
#pragma unroll
      for (int n = 0; n < 8; ++n) acc[m][n] = vzero;

    if (t > 0) {
      // ---- GEMM: acc[m][n] += h(64x256,bf16) @ W_hh^T tile, software pipelined ----
      v16bf aA[2][4];
      v16bf bb[4];
      loadA(aA[0], 0);
#pragma unroll
      for (int s = 0; s < 4; ++s) bb[s] = loadB(0, s);   // fragments s=0..3 (kt=0)
#pragma unroll
      for (int kt = 0; kt < 8; ++kt) {
        if (kt + 1 < 8) loadA(aA[(kt + 1) & 1], kt + 1); // prefetch next A set
#pragma unroll
        for (int n = 0; n < 8; ++n) {
          const int s = kt * 8 + n;
          v16bf bcur = bb[s & 3];
          if (s + 4 < 64) bb[s & 3] = loadB((s + 4) >> 3, (s + 4) & 7);
#pragma unroll
          for (int m = 0; m < 4; ++m) {
            acc[m][n] = __builtin_amdgcn_wmma_f32_16x16x32_bf16(
                false, aA[kt & 1][m], false, bcur, (short)0, acc[m][n], false, false);
          }
        }
      }
    }
    __syncthreads();   // all A-fragment LDS reads complete before h is overwritten

    // ---- LSTM cell update directly on accumulator layout ----
#pragma unroll
    for (int m = 0; m < 4; ++m) {
#pragma unroll
      for (int r = 0; r < 8; ++r) {
        const int row = m * 16 + rhi + r;
        const int nid = nodeids[(row0 + row) * PLEN + t];
        const float* xgb = ngates + (size_t)nid * GDIM + dbase;
#pragma unroll
        for (int j = 0; j < 2; ++j) {
          float gi = xgb[j * 16 +   0] + acc[m][0 * 2 + j][r];
          float gf = xgb[j * 16 + 256] + acc[m][1 * 2 + j][r];
          float gg = xgb[j * 16 + 512] + acc[m][2 * 2 + j][r];
          float go = xgb[j * 16 + 768] + acc[m][3 * 2 + j][r];
          const int ci = (m * 2 + j) * 8 + r;
          float cn = sigm(gf) * c[ci] + sigm(gi) * tanh_f(gg);
          c[ci] = cn;
          float ht = sigm(go) * tanh_f(cn);
          hls[(size_t)row * HST + dbase + j * 16] = f2bf(ht);
          if (t == PLEN - 1) out[(size_t)(row0 + row) * DD + dbase + j * 16] = ht;
        }
      }
    }
    __syncthreads();   // new h visible before next step's A-fragment loads
  }
}

// -----------------------------------------------------------------------------
extern "C" void kernel_launch(void* const* d_in, const int* in_sizes, int n_in,
                              void* d_out, int out_size, void* d_ws, size_t ws_size,
                              hipStream_t stream) {
  (void)in_sizes; (void)n_in; (void)out_size; (void)ws_size;
  const float* cross = (const float*)d_in[0];
  const float* emb   = (const float*)d_in[1];
  const float* Wih   = (const float*)d_in[2];
  const float* Whh   = (const float*)d_in[3];
  const float* bih   = (const float*)d_in[4];
  const float* bhh   = (const float*)d_in[5];
  float* out = (float*)d_out;

  char* ws = (char*)d_ws;
  unsigned short* packW  = (unsigned short*)ws;
  float*          ngates = (float*)(ws + NGATES_OFF);
  int*            nids   = (int*)(ws + NIDS_OFF);

  hipLaunchKernelGGL(pack_whh_kernel, dim3(512), dim3(32), 0, stream, Whh, packW);
  hipLaunchKernelGGL(leaf_paths_kernel, dim3((NROWS + 255) / 256), dim3(256), 0, stream,
                     cross, nids);
  hipLaunchKernelGGL(node_gates_kernel, dim3(8, 39), dim3(128), 0, stream,
                     emb, Wih, bih, bhh, ngates);
  hipLaunchKernelGGL(lstm_fused_kernel, dim3(NROWS / MROWS), dim3(256), 0, stream,
                     packW, ngates, nids, out);
}
